// AICSD_model_5634997092911
// MI455X (gfx1250) — compile-verified, matrix operand order checked by reading, not run on hardware
//
#include <hip/hip_runtime.h>
#include <hip/hip_bf16.h>

typedef __attribute__((ext_vector_type(16))) _Float16 v16h;
typedef __attribute__((ext_vector_type(8)))  _Float16 v8h;
typedef __attribute__((ext_vector_type(8)))  float    v8f;

#define NB      8          // batch
#define N19     19
#define SPOS    (NB*65*65) // 33800 output positions of the 1x1 conv
#define NTILES  ((SPOS+15)/16)   // 2113
#define KROW    16641      // 129*129
#define KPAD    16672      // rounded up to multiple of 32
#define ROWS32  32         // padded channel rows per batch for WMMA gram
#define GCHUNK  2112       // k-chunk per gram workgroup (66 * 32)
#define NCHUNK  8          // ceil(16672/2112)
#define NSLOT   (2*NB*NCHUNK)  // 128 gram partial slots

__device__ __forceinline__ v8f wmma16(v16h a, v16h b, v8f c) {
    return __builtin_amdgcn_wmma_f32_16x16x32_f16(false, a, false, b, (short)0, c,
                                                  false, false);
}

__device__ __forceinline__ v16h load_frag(const _Float16* base, int koff) {
    v8h lo = *(const v8h*)(base + koff);
    v8h hi = *(const v8h*)(base + 16 + koff);
    v16h t;
#pragma unroll
    for (int i = 0; i < 8; ++i) { t[i] = lo[i]; t[8 + i] = hi[i]; }
    return t;
}

// ---------------------------------------------------------------- init
__global__ __launch_bounds__(256) void k_init(unsigned* ps_u32, unsigned* pt_u32,
                                              float* partials,
                                              const float* convw, _Float16* convw16) {
    long stride = (long)gridDim.x * blockDim.x;
    long tid = (long)blockIdx.x * blockDim.x + threadIdx.x;
    const long NZ = (long)NB * ROWS32 * KPAD / 2;  // f16 pairs
    for (long i = tid; i < NZ; i += stride) { ps_u32[i] = 0u; pt_u32[i] = 0u; }
    for (long i = tid; i < 4096; i += stride) partials[i] = 0.f;
    for (long i = tid; i < 768 * 128; i += stride) convw16[i] = (_Float16)convw[i];
}

// ----------------------------------------------- row softmax of s_logit
__global__ __launch_bounds__(256) void k_softmax_s(const float* __restrict__ s_logit,
                                                   _Float16* __restrict__ ps,
                                                   float* __restrict__ Erows) {
    __shared__ float red[256];
    int row = blockIdx.x;                 // 0..151 = b*19+n
    int b = row / N19, n = row % N19;
    const float* src = s_logit + (size_t)row * KROW;
    _Float16* dst = ps + (size_t)(b * ROWS32 + n) * KPAD;
    int tid = threadIdx.x;

    float m = -1e30f;
    for (int k = tid; k < KROW; k += 256) m = fmaxf(m, src[k]);
    red[tid] = m; __syncthreads();
    for (int s = 128; s > 0; s >>= 1) { if (tid < s) red[tid] = fmaxf(red[tid], red[tid + s]); __syncthreads(); }
    m = red[0]; __syncthreads();

    float sum = 0.f;
    for (int k = tid; k < KROW; k += 256) sum += __expf(src[k] - m);
    red[tid] = sum; __syncthreads();
    for (int s = 128; s > 0; s >>= 1) { if (tid < s) red[tid] += red[tid + s]; __syncthreads(); }
    sum = red[0]; __syncthreads();

    float logS = __logf(sum), inv = 1.f / sum;
    float eacc = 0.f;
    for (int k = tid; k < KROW; k += 256) {
        float lx = src[k] - m;
        float p = __expf(lx) * inv;
        dst[k] = (_Float16)(p * 1024.0f);   // scale into healthy f16 range
        eacc += p * (lx - logS);            // p * log p
    }
    if (tid < (KPAD - KROW)) dst[KROW + tid] = (_Float16)0.f;
    red[tid] = eacc; __syncthreads();
    for (int s = 128; s > 0; s >>= 1) { if (tid < s) red[tid] += red[tid + s]; __syncthreads(); }
    if (tid == 0) Erows[row] = red[0];
}

// ---------------- row softmax of bilinear-resized t_logit (65x65 -> 129x129)
__global__ __launch_bounds__(256) void k_softmax_t(const float* __restrict__ t_logit,
                                                   _Float16* __restrict__ pt,
                                                   float* __restrict__ Erows) {
    __shared__ float srcrow[65 * 65];
    __shared__ float red[256];
    int row = blockIdx.x;
    int b = row / N19, n = row % N19;
    const float* src = t_logit + (size_t)row * (65 * 65);
    _Float16* dst = pt + (size_t)(b * ROWS32 + n) * KPAD;
    int tid = threadIdx.x;
    for (int k = tid; k < 65 * 65; k += 256) srcrow[k] = src[k];
    __syncthreads();

    auto resv = [&](int k) -> float {      // align_corners bilinear, scale exactly 0.5
        int y = k / 129, x = k - y * 129;
        int y0 = y >> 1, x0 = x >> 1;
        int y1 = min(y0 + 1, 64), x1 = min(x0 + 1, 64);
        float wy = (y & 1) ? 0.5f : 0.f;
        float wx = (x & 1) ? 0.5f : 0.f;
        float r0 = srcrow[y0 * 65 + x0] * (1.f - wx) + srcrow[y0 * 65 + x1] * wx;
        float r1 = srcrow[y1 * 65 + x0] * (1.f - wx) + srcrow[y1 * 65 + x1] * wx;
        return r0 * (1.f - wy) + r1 * wy;
    };

    float m = -1e30f;
    for (int k = tid; k < KROW; k += 256) m = fmaxf(m, resv(k));
    red[tid] = m; __syncthreads();
    for (int s = 128; s > 0; s >>= 1) { if (tid < s) red[tid] = fmaxf(red[tid], red[tid + s]); __syncthreads(); }
    m = red[0]; __syncthreads();

    float sum = 0.f;
    for (int k = tid; k < KROW; k += 256) sum += __expf(resv(k) - m);
    red[tid] = sum; __syncthreads();
    for (int s = 128; s > 0; s >>= 1) { if (tid < s) red[tid] += red[tid + s]; __syncthreads(); }
    sum = red[0]; __syncthreads();

    float logS = __logf(sum), inv = 1.f / sum;
    float eacc = 0.f;
    for (int k = tid; k < KROW; k += 256) {
        float lx = resv(k) - m;
        float p = __expf(lx) * inv;
        dst[k] = (_Float16)(p * 1024.0f);
        eacc += p * (lx - logS);
    }
    if (tid < (KPAD - KROW)) dst[KROW + tid] = (_Float16)0.f;
    red[tid] = eacc; __syncthreads();
    for (int s = 128; s > 0; s >>= 1) { if (tid < s) red[tid] += red[tid + s]; __syncthreads(); }
    if (tid == 0) Erows[152 + row] = red[0];
}

// ------- fused: bilinear resize + 1x1-conv WMMA GEMM + dual softmax + KL
// one block = 16 spatial positions; 8 waves each own 6 N-tiles of 16 channels
__global__ __launch_bounds__(256) void k_conv_kl(const float* __restrict__ s_out,
                                                 const float* __restrict__ t_out,
                                                 const _Float16* __restrict__ w16,
                                                 const float* __restrict__ bias,
                                                 float* __restrict__ partials) {
    extern __shared__ char smem[];
    _Float16* A = (_Float16*)smem;                 // [16][136] f16 (pad kills bank conflicts)
    float* Z = (float*)(smem + 16 * 136 * 2);      // [16][772] conv output
    float* T = Z + 16 * 772;                       // [16][772] t_out tile
    __shared__ int pb[16], py[16], px[16], pbase[16];
    __shared__ float pvalid[16];
    __shared__ float possum[16];

    int tile = blockIdx.x, tid = threadIdx.x;
    if (tid < 16) {
        long gp = (long)tile * 16 + tid;
        int valid = gp < SPOS;
        long g = valid ? gp : 0;
        int b = (int)(g / 4225);
        int rem = (int)(g - (long)b * 4225);
        int y = rem / 65, x = rem - y * 65;
        pb[tid] = b; py[tid] = y; px[tid] = x;
        pbase[tid] = (b * 768) * 4225 + y * 65 + x;    // < 2^31
        pvalid[tid] = valid ? 1.f : 0.f;
    }
    __syncthreads();

    // A tile: on-the-fly bilinear resize of s_out (33->65, weights {0,0.5}), cast f16
    for (int e = tid; e < 16 * 128; e += 256) {
        int p = e & 15, c = e >> 4;
        int b = pb[p], y = py[p], x = px[p];
        int y0 = y >> 1, x0 = x >> 1;
        int y1 = min(y0 + 1, 32), x1 = min(x0 + 1, 32);
        float wy = (y & 1) ? 0.5f : 0.f;
        float wx = (x & 1) ? 0.5f : 0.f;
        const float* s = s_out + (size_t)(b * 128 + c) * 1089;
        float r0 = s[y0 * 33 + x0] * (1.f - wx) + s[y0 * 33 + x1] * wx;
        float r1 = s[y1 * 33 + x0] * (1.f - wx) + s[y1 * 33 + x1] * wx;
        A[p * 136 + c] = (_Float16)(r0 * (1.f - wy) + r1 * wy);
    }
    // T tile: coalesced (16 consecutive positions per channel)
    for (int e = tid; e < 16 * 768; e += 256) {
        int p = e & 15, c = e >> 4;
        T[p * 772 + c] = t_out[(size_t)pbase[p] + (size_t)c * 4225];
    }
    __syncthreads();

    // GEMM: M=16 positions, K=128 (4 wmma k-steps), N=768 (48 tiles / 8 waves)
    int wave = tid >> 5, lane = tid & 31;
    int mrow = lane & 15;
    int koff = (lane < 16) ? 0 : 8;    // 16-bit A/B lane K split
    v16h a[4];
#pragma unroll
    for (int ks = 0; ks < 4; ++ks)
        a[ks] = load_frag(A + mrow * 136 + ks * 32, koff);

    int ntg0 = wave * 6;
    v16h bt[4], bn[4];
    {   // preload first N-tile's weight fragments (one clause of 8 b128 loads)
        const _Float16* wrow = w16 + (size_t)(ntg0 * 16 + mrow) * 128;
#pragma unroll
        for (int ks = 0; ks < 4; ++ks) bt[ks] = load_frag(wrow + ks * 32, koff);
    }
#pragma unroll
    for (int j = 0; j < 6; ++j) {
        int cc = (ntg0 + j) * 16 + mrow;             // output channel (B column)
        if (j < 5) {                                 // prefetch next tile before MACs
            const _Float16* wn = w16 + (size_t)((ntg0 + j + 1) * 16 + mrow) * 128;
#pragma unroll
            for (int ks = 0; ks < 4; ++ks) bn[ks] = load_frag(wn + ks * 32, koff);
        }
        float bv = bias[cc];
        v8f acc = {};
#pragma unroll
        for (int ks = 0; ks < 4; ++ks) acc = wmma16(a[ks], bt[ks], acc);
        int mb = (lane >> 4) << 3;
#pragma unroll
        for (int r = 0; r < 8; ++r) Z[(mb + r) * 772 + cc] = acc[r] + bv;
        if (j < 5) {
#pragma unroll
            for (int ks = 0; ks < 4; ++ks) bt[ks] = bn[ks];
        }
    }
    __syncthreads();

    // per-position: channel softmax of Z and T, KL contribution
    for (int pi = 0; pi < 2; ++pi) {
        int p = wave * 2 + pi;
        const float* zr = Z + p * 772;
        const float* tr = T + p * 772;
        float mz = -1e30f, mt = -1e30f;
        for (int c = lane; c < 768; c += 32) { mz = fmaxf(mz, zr[c]); mt = fmaxf(mt, tr[c]); }
        for (int o = 16; o; o >>= 1) {
            mz = fmaxf(mz, __shfl_xor(mz, o, 32));
            mt = fmaxf(mt, __shfl_xor(mt, o, 32));
        }
        float sz = 0.f, st = 0.f;
        for (int c = lane; c < 768; c += 32) { sz += __expf(zr[c] - mz); st += __expf(tr[c] - mt); }
        for (int o = 16; o; o >>= 1) { sz += __shfl_xor(sz, o, 32); st += __shfl_xor(st, o, 32); }
        float lzs = mz + __logf(sz), lzt = mt + __logf(st);
        float acc = 0.f;
        for (int c = lane; c < 768; c += 32) {
            float tv = tr[c];
            float lpt = tv - lzt;
            acc += __expf(lpt) * (lpt - (zr[c] - lzs));
        }
        for (int o = 16; o; o >>= 1) acc += __shfl_xor(acc, o, 32);
        if (lane == 0) possum[p] = acc * pvalid[p];
    }
    __syncthreads();
    if (tid == 0) {
        float s = 0.f;
        for (int p = 0; p < 16; ++p) s += possum[p];   // fixed order: deterministic
        partials[tile] = s;
    }
}

// ------------- Gram partials G = P * P^T via WMMA (same fragment = A and B)
__global__ __launch_bounds__(32) void k_gram(const _Float16* __restrict__ ps,
                                             const _Float16* __restrict__ pt,
                                             float* __restrict__ gslots) {
    int slot = blockIdx.x;              // mat*64 + b*8 + chunk
    int mat = slot >> 6;
    int rest = slot & 63;
    int b = rest >> 3, chunk = rest & 7;
    const _Float16* P = (mat ? pt : ps) + (size_t)b * ROWS32 * KPAD;
    int lane = threadIdx.x;
    int koff = (lane < 16) ? 0 : 8;
    int rlo = lane & 15;
    v8f a00 = {}, a01 = {}, a10 = {}, a11 = {};
    int k0 = chunk * GCHUNK;
    int k1 = min(k0 + GCHUNK, KPAD);
    for (int k = k0; k < k1; k += 64) {   // unroll x2: more loads in flight per wmma wait
        v16h f0 = load_frag(P + (size_t)rlo * KPAD + k, koff);
        v16h f1 = load_frag(P + (size_t)(rlo + 16) * KPAD + k, koff);
        v16h g0 = load_frag(P + (size_t)rlo * KPAD + k + 32, koff);
        v16h g1 = load_frag(P + (size_t)(rlo + 16) * KPAD + k + 32, koff);
        a00 = wmma16(f0, f0, a00);
        a01 = wmma16(f0, f1, a01);
        a10 = wmma16(f1, f0, a10);
        a11 = wmma16(f1, f1, a11);
        a00 = wmma16(g0, g0, a00);
        a01 = wmma16(g0, g1, a01);
        a10 = wmma16(g1, g0, a10);
        a11 = wmma16(g1, g1, a11);
    }
    float* out = gslots + (size_t)slot * 1024;   // 32x32
    int n = lane & 15, mb = (lane >> 4) << 3;
#pragma unroll
    for (int r = 0; r < 8; ++r) {
        out[(mb + r) * 32 + n]            = a00[r];
        out[(mb + r) * 32 + 16 + n]       = a01[r];
        out[(16 + mb + r) * 32 + n]       = a10[r];
        out[(16 + mb + r) * 32 + 16 + n]  = a11[r];
    }
}

// -------------------------------- finalize: ICC algebra + both reductions
__global__ __launch_bounds__(512) void k_final(const float* __restrict__ gslots,
                                               const float* __restrict__ Erows,
                                               const float* __restrict__ partials,
                                               float* __restrict__ out) {
    __shared__ float Gs[361], Gt[361], Es[19], Et[19], nrs[19], nrt[19];
    __shared__ float red[512];
    int tid = threadIdx.x;
    if (tid < 361) {
        int i = tid / 19, j = tid % 19;
        int lo = min(i, j), hi = max(i, j);
        float gs = 0.f, gt = 0.f;
        for (int s = 0; s < 64; ++s)  gs += gslots[(size_t)s * 1024 + lo * 32 + hi];
        for (int s = 64; s < 128; ++s) gt += gslots[(size_t)s * 1024 + lo * 32 + hi];
        Gs[tid] = gs * (1.0f / 1048576.0f);    // undo 2^10 * 2^10 scaling
        Gt[tid] = gt * (1.0f / 1048576.0f);
    }
    if (tid < 19) { float e = 0.f; for (int b = 0; b < 8; ++b) e += Erows[b * 19 + tid]; Es[tid] = e; }
    if (tid >= 32 && tid < 51) {
        int c = tid - 32; float e = 0.f;
        for (int b = 0; b < 8; ++b) e += Erows[152 + b * 19 + c];
        Et[c] = e;
    }
    __syncthreads();
    if (tid < 361) {                       // sym[i,j] = (E[max] - G[min,max]) / 8
        int i = tid / 19, j = tid % 19;
        int hi = max(i, j);
        Gs[tid] = (Es[hi] - Gs[tid]) * 0.125f;
        Gt[tid] = (Et[hi] - Gt[tid]) * 0.125f;
    }
    __syncthreads();
    if (tid < 19) {
        float a = 0.f, b2 = 0.f;
        for (int j = 0; j < 19; ++j) {
            float v = Gs[tid * 19 + j]; a += v * v;
            v = Gt[tid * 19 + j]; b2 += v * v;
        }
        nrs[tid] = fmaxf(sqrtf(a), 1e-12f);
        nrt[tid] = fmaxf(sqrtf(b2), 1e-12f);
    }
    __syncthreads();
    float part = 0.f;
    if (tid < 361) {
        int i = tid / 19;
        float d = Gs[tid] / nrs[i] - Gt[tid] / nrt[i];
        part = d * d;
    }
    red[tid] = part; __syncthreads();
    for (int s = 256; s > 0; s >>= 1) { if (tid < s) red[tid] += red[tid + s]; __syncthreads(); }
    if (tid == 0) out[1] = red[0];
    __syncthreads();
    float psum = 0.f;
    for (int t = tid; t < 4096; t += 512) psum += partials[t];   // zero-padded, fixed order
    red[tid] = psum; __syncthreads();
    for (int s = 256; s > 0; s >>= 1) { if (tid < s) red[tid] += red[tid + s]; __syncthreads(); }
    if (tid == 0) out[0] = red[0] * (1.0f / 25958400.0f);        // mean over B*C*H*W
}

extern "C" void kernel_launch(void* const* d_in, const int* in_sizes, int n_in,
                              void* d_out, int out_size, void* d_ws, size_t ws_size,
                              hipStream_t stream) {
    (void)in_sizes; (void)n_in; (void)out_size; (void)ws_size;
    const float* s_out   = (const float*)d_in[0];
    const float* t_out   = (const float*)d_in[1];
    const float* t_logit = (const float*)d_in[2];
    const float* s_logit = (const float*)d_in[3];
    const float* conv_w  = (const float*)d_in[4];
    const float* conv_b  = (const float*)d_in[5];
    float* out = (float*)d_out;

    char* ws = (char*)d_ws;
    size_t off = 0;
    auto carve = [&](size_t bytes) -> char* {
        char* r = ws + off;
        off += (bytes + 255) & ~(size_t)255;
        return r;
    };
    _Float16* w16     = (_Float16*)carve((size_t)768 * 128 * 2);
    _Float16* ps      = (_Float16*)carve((size_t)NB * ROWS32 * KPAD * 2);
    _Float16* pt      = (_Float16*)carve((size_t)NB * ROWS32 * KPAD * 2);
    float*    Erows   = (float*)carve(512 * 4);
    float*    gslots  = (float*)carve((size_t)NSLOT * 1024 * 4);
    float*    parts   = (float*)carve(4096 * 4);

    k_init<<<1024, 256, 0, stream>>>((unsigned*)ps, (unsigned*)pt, parts, conv_w, w16);
    k_softmax_s<<<152, 256, 0, stream>>>(s_logit, ps, Erows);
    k_softmax_t<<<152, 256, 0, stream>>>(t_logit, pt, Erows);
    size_t smem = (size_t)16 * 136 * 2 + (size_t)2 * 16 * 772 * 4;  // 103168 B
    k_conv_kl<<<NTILES, 256, smem, stream>>>(s_out, t_out, w16, conv_b, parts);
    k_gram<<<NSLOT, 32, 0, stream>>>(ps, pt, gslots);
    k_final<<<1, 512, 0, stream>>>(gslots, Erows, parts, out);
}
